// FeedForwardBlock_74749610820169
// MI455X (gfx1250) — matrix operand order repaired and checked by reading, not compiled, not run on hardware
//
#include <hip/hip_runtime.h>

// MI455X / gfx1250, wave32. Complex GEMMs via v_wmma_f32_16x16x32_bf16,
// fp32 accumulate, split-K for occupancy, double-buffered LDS A-staging,
// non-temporal streaming of the 512MB U0 operand.
typedef __attribute__((ext_vector_type(16))) __bf16 v16bf;
typedef __attribute__((ext_vector_type(8)))  __bf16 v8bf;
typedef __attribute__((ext_vector_type(8)))  float  v8f;

#define B_   64
#define D_   8192
#define K_   16
#define DR_  512
#define KC   128                 // K-chunk staged in LDS
#define LSTR 136                 // 128 + 8 bf16 pad -> conflict-free frag reads
#define LBUF (64 * LSTR)         // bf16 per buffer per component (double-buffered)

// ---------------- Kernel 1: normalize psi rows, convert to bf16 --------------
__global__ void k_normalize(const float* __restrict__ pre,
                            const float* __restrict__ pim,
                            __bf16* __restrict__ bre,
                            __bf16* __restrict__ bim) {
    __shared__ float red[256];
    const int row = blockIdx.x;
    const int tid = threadIdx.x;
    const float* r = pre + row * D_;
    const float* m = pim + row * D_;
    float s = 0.f;
    for (int i = tid; i < D_; i += 256) {
        float a = r[i], b = m[i];
        s += a * a + b * b;
    }
    red[tid] = s;
    __syncthreads();
    for (int off = 128; off > 0; off >>= 1) {
        if (tid < off) red[tid] += red[tid + off];
        __syncthreads();
    }
    const float inv = rsqrtf(red[0] + 1e-12f);
    __bf16* orow = bre + row * D_;
    __bf16* irow = bim + row * D_;
    for (int i = tid; i < D_; i += 256) {
        orow[i] = (__bf16)(r[i] * inv);
        irow[i] = (__bf16)(m[i] * inv);
    }
}

// ---------------- Kernel 2: psi1 partials = psi_n @ U0^T (split-K) -----------
// grid (32, S1): x -> 16 N-tiles per WG (one per wave), y -> K segment.
__global__ __launch_bounds__(512)
void k_gemm1(const __bf16* __restrict__ are,
             const __bf16* __restrict__ aim,
             const float* __restrict__ U0re,
             const float* __restrict__ U0im,
             float* __restrict__ pre,          // [S1][64][D_]
             float* __restrict__ pim,
             int seg_len) {
    __shared__ __bf16 lre[2 * LBUF];
    __shared__ __bf16 lim[2 * LBUF];
    const int tid  = threadIdx.x;
    const int lane = tid & 31;
    const int wave = tid >> 5;
    const int hi   = lane >> 4;
    const int lmod = lane & 15;
    const int n0   = (blockIdx.x * 16 + wave) * 16;
    const int seg  = blockIdx.y;
    const int kbeg = seg * seg_len;
    const int nchunks = seg_len / KC;

    v8f acc_re[4], acc_im[4];
#pragma unroll
    for (int mt = 0; mt < 4; ++mt)
#pragma unroll
        for (int i = 0; i < 8; ++i) { acc_re[mt][i] = 0.f; acc_im[mt][i] = 0.f; }

    uint2 sR[4], sI[4];
    // prologue: stage chunk 0 -> buffer 0
#pragma unroll
    for (int it = 0; it < 4; ++it) {
        const int id = tid + it * 512, rr = id >> 5, kg = id & 31;
        sR[it] = *(const uint2*)(are + rr * D_ + kbeg + kg * 4);
        sI[it] = *(const uint2*)(aim + rr * D_ + kbeg + kg * 4);
    }
#pragma unroll
    for (int it = 0; it < 4; ++it) {
        const int id = tid + it * 512, rr = id >> 5, kg = id & 31;
        *(uint2*)(lre + rr * LSTR + kg * 4) = sR[it];
        *(uint2*)(lim + rr * LSTR + kg * 4) = sI[it];
    }
    __syncthreads();

    for (int c = 0; c < nchunks; ++c) {
        const int cur  = (c & 1) * LBUF;
        const int nxt  = LBUF - cur;
        const bool more = (c + 1) < nchunks;
        if (more) {                         // prefetch next chunk into registers
            const int cc = kbeg + (c + 1) * KC;
#pragma unroll
            for (int it = 0; it < 4; ++it) {
                const int id = tid + it * 512, rr = id >> 5, kg = id & 31;
                sR[it] = *(const uint2*)(are + rr * D_ + cc + kg * 4);
                sI[it] = *(const uint2*)(aim + rr * D_ + cc + kg * 4);
            }
        }
        const int kb0 = kbeg + c * KC;
#pragma unroll
        for (int ks = 0; ks < KC / 32; ++ks) {
            const int kb = kb0 + ks * 32;
            v16bf bre, bim, bimn;          // B frag: lane%16 = N col, elements = K (table order)
#pragma unroll
            for (int h = 0; h < 16; ++h) {
                const size_t gi = (size_t)(n0 + lmod) * D_ + kb + 8 * hi + h + (h & 8);
                const float vr = __builtin_nontemporal_load(U0re + gi);
                const float vi = __builtin_nontemporal_load(U0im + gi);
                bre[h]  = (__bf16)vr;
                bim[h]  = (__bf16)vi;
                bimn[h] = (__bf16)(-vi);
            }
            const int kl = ks * 32 + 8 * hi;
#pragma unroll
            for (int mt = 0; mt < 4; ++mt) {
                const int rbase = cur + (16 * mt + lmod) * LSTR;
                const v8bf relo = *(const v8bf*)(lre + rbase + kl);
                const v8bf rehi = *(const v8bf*)(lre + rbase + kl + 16);
                const v8bf imlo = *(const v8bf*)(lim + rbase + kl);
                const v8bf imhi = *(const v8bf*)(lim + rbase + kl + 16);
                v16bf aR, aI;
#pragma unroll
                for (int i = 0; i < 8; ++i) {
                    aR[i] = relo[i]; aR[i + 8] = rehi[i];
                    aI[i] = imlo[i]; aI[i + 8] = imhi[i];
                }
                acc_re[mt] = __builtin_amdgcn_wmma_f32_16x16x32_bf16(false, aR, false, bre,  (short)0, acc_re[mt], false, false);
                acc_re[mt] = __builtin_amdgcn_wmma_f32_16x16x32_bf16(false, aI, false, bimn, (short)0, acc_re[mt], false, false);
                acc_im[mt] = __builtin_amdgcn_wmma_f32_16x16x32_bf16(false, aR, false, bim,  (short)0, acc_im[mt], false, false);
                acc_im[mt] = __builtin_amdgcn_wmma_f32_16x16x32_bf16(false, aI, false, bre,  (short)0, acc_im[mt], false, false);
            }
        }
        if (more) {                         // commit staged regs to other buffer
#pragma unroll
            for (int it = 0; it < 4; ++it) {
                const int id = tid + it * 512, rr = id >> 5, kg = id & 31;
                *(uint2*)(lre + nxt + rr * LSTR + kg * 4) = sR[it];
                *(uint2*)(lim + nxt + rr * LSTR + kg * 4) = sI[it];
            }
        }
        __syncthreads();
    }

    float* po_re = pre + (size_t)seg * (B_ * D_);
    float* po_im = pim + (size_t)seg * (B_ * D_);
#pragma unroll
    for (int mt = 0; mt < 4; ++mt)
#pragma unroll
        for (int r = 0; r < 8; ++r) {
            const int row = 16 * mt + r + 8 * hi;
            const int col = n0 + lmod;
            po_re[row * D_ + col] = acc_re[mt][r];
            po_im[row * D_ + col] = acc_im[mt][r];
        }
}

// ---------------- Kernel 3: reduce partials -> w, a_n (bf16) -----------------
__global__ void k_stats(const float* __restrict__ pre,
                        const float* __restrict__ pim,
                        int nseg,
                        float* __restrict__ wbuf,
                        __bf16* __restrict__ anre,
                        __bf16* __restrict__ anim) {
    const int lane = threadIdx.x & 31;
    const int task = blockIdx.x * 8 + (threadIdx.x >> 5);  // b*16 + kout
    const int base = task * DR_;
    float re[16], im[16];
#pragma unroll
    for (int i = 0; i < 16; ++i) { re[i] = 0.f; im[i] = 0.f; }
    for (int sg = 0; sg < nseg; ++sg) {
        const size_t off = (size_t)sg * (B_ * D_) + base;
#pragma unroll
        for (int i = 0; i < 16; ++i) {
            const int d = lane + 32 * i;
            re[i] += pre[off + d];
            im[i] += pim[off + d];
        }
    }
    float s = 0.f;
#pragma unroll
    for (int i = 0; i < 16; ++i) s += re[i] * re[i] + im[i] * im[i];
#pragma unroll
    for (int m = 16; m > 0; m >>= 1) s += __shfl_xor(s, m, 32);
    const float inv = rsqrtf(s + 1e-12f);
#pragma unroll
    for (int i = 0; i < 16; ++i) {
        const int d = base + lane + 32 * i;
        anre[d] = (__bf16)(re[i] * inv);
        anim[d] = (__bf16)(im[i] * inv);
    }
    if (lane == 0) wbuf[task] = s;
}

// ---------------- Kernel 4: expert GEMM partials (split-K) -------------------
// grid (32, S2): x -> (kout, half), y -> K segment of DR.
__global__ __launch_bounds__(512)
void k_gemm2(const __bf16* __restrict__ are,
             const __bf16* __restrict__ aim,
             const float* __restrict__ Vre,
             const float* __restrict__ Vim,
             const int* __restrict__ eids,
             float* __restrict__ pre,          // [S2][64][D_]
             float* __restrict__ pim,
             int seg_len) {
    __shared__ __bf16 lre[2 * LBUF];
    __shared__ __bf16 lim[2 * LBUF];
    const int tid  = threadIdx.x;
    const int lane = tid & 31;
    const int wave = tid >> 5;
    const int hi   = lane >> 4;
    const int lmod = lane & 15;
    const int kout = blockIdx.x >> 1;
    const int half = blockIdx.x & 1;
    const int n0   = (half * 16 + wave) * 16;
    const int seg  = blockIdx.y;
    const int kbeg = seg * seg_len;
    const int nchunks = seg_len / KC;
    const size_t vbase = (size_t)eids[kout] * DR_ * DR_;
    const int abase = kout * DR_;

    v8f acc_re[4], acc_im[4];
#pragma unroll
    for (int mt = 0; mt < 4; ++mt)
#pragma unroll
        for (int i = 0; i < 8; ++i) { acc_re[mt][i] = 0.f; acc_im[mt][i] = 0.f; }

    uint2 sR[4], sI[4];
#pragma unroll
    for (int it = 0; it < 4; ++it) {
        const int id = tid + it * 512, rr = id >> 5, kg = id & 31;
        sR[it] = *(const uint2*)(are + rr * D_ + abase + kbeg + kg * 4);
        sI[it] = *(const uint2*)(aim + rr * D_ + abase + kbeg + kg * 4);
    }
#pragma unroll
    for (int it = 0; it < 4; ++it) {
        const int id = tid + it * 512, rr = id >> 5, kg = id & 31;
        *(uint2*)(lre + rr * LSTR + kg * 4) = sR[it];
        *(uint2*)(lim + rr * LSTR + kg * 4) = sI[it];
    }
    __syncthreads();

    for (int c = 0; c < nchunks; ++c) {
        const int cur  = (c & 1) * LBUF;
        const int nxt  = LBUF - cur;
        const bool more = (c + 1) < nchunks;
        if (more) {
            const int cc = abase + kbeg + (c + 1) * KC;
#pragma unroll
            for (int it = 0; it < 4; ++it) {
                const int id = tid + it * 512, rr = id >> 5, kg = id & 31;
                sR[it] = *(const uint2*)(are + rr * D_ + cc + kg * 4);
                sI[it] = *(const uint2*)(aim + rr * D_ + cc + kg * 4);
            }
        }
        const int kb0 = kbeg + c * KC;
#pragma unroll
        for (int ks = 0; ks < KC / 32; ++ks) {
            const int kb = kb0 + ks * 32;
            v16bf bre, bim, bimn;
#pragma unroll
            for (int h = 0; h < 16; ++h) {
                const size_t gi = vbase + (size_t)(n0 + lmod) * DR_ + kb + 8 * hi + h + (h & 8);
                const float vr = Vre[gi];
                const float vi = Vim[gi];
                bre[h]  = (__bf16)vr;
                bim[h]  = (__bf16)vi;
                bimn[h] = (__bf16)(-vi);
            }
            const int kl = ks * 32 + 8 * hi;
#pragma unroll
            for (int mt = 0; mt < 4; ++mt) {
                const int rbase = cur + (16 * mt + lmod) * LSTR;
                const v8bf relo = *(const v8bf*)(lre + rbase + kl);
                const v8bf rehi = *(const v8bf*)(lre + rbase + kl + 16);
                const v8bf imlo = *(const v8bf*)(lim + rbase + kl);
                const v8bf imhi = *(const v8bf*)(lim + rbase + kl + 16);
                v16bf aR, aI;
#pragma unroll
                for (int i = 0; i < 8; ++i) {
                    aR[i] = relo[i]; aR[i + 8] = rehi[i];
                    aI[i] = imlo[i]; aI[i + 8] = imhi[i];
                }
                acc_re[mt] = __builtin_amdgcn_wmma_f32_16x16x32_bf16(false, aR, false, bre,  (short)0, acc_re[mt], false, false);
                acc_re[mt] = __builtin_amdgcn_wmma_f32_16x16x32_bf16(false, aI, false, bimn, (short)0, acc_re[mt], false, false);
                acc_im[mt] = __builtin_amdgcn_wmma_f32_16x16x32_bf16(false, aR, false, bim,  (short)0, acc_im[mt], false, false);
                acc_im[mt] = __builtin_amdgcn_wmma_f32_16x16x32_bf16(false, aI, false, bre,  (short)0, acc_im[mt], false, false);
            }
        }
        if (more) {
#pragma unroll
            for (int it = 0; it < 4; ++it) {
                const int id = tid + it * 512, rr = id >> 5, kg = id & 31;
                *(uint2*)(lre + nxt + rr * LSTR + kg * 4) = sR[it];
                *(uint2*)(lim + nxt + rr * LSTR + kg * 4) = sI[it];
            }
        }
        __syncthreads();
    }

    float* po_re = pre + (size_t)seg * (B_ * D_);
    float* po_im = pim + (size_t)seg * (B_ * D_);
#pragma unroll
    for (int mt = 0; mt < 4; ++mt)
#pragma unroll
        for (int r = 0; r < 8; ++r) {
            const int row = 16 * mt + r + 8 * hi;
            const int col = kout * DR_ + n0 + lmod;
            po_re[row * D_ + col] = acc_re[mt][r];
            po_im[row * D_ + col] = acc_im[mt][r];
        }
}

// ---------------- Kernel 5: combine partials -> probs ------------------------
__global__ void k_out(const float* __restrict__ pre,
                      const float* __restrict__ pim,
                      int nseg,
                      const float* __restrict__ wbuf,
                      float* __restrict__ out) {
    const int idx = blockIdx.x * 256 + threadIdx.x;       // 0 .. B_*D_-1
    float cr = 0.f, ci = 0.f;
    for (int sg = 0; sg < nseg; ++sg) {
        cr += pre[(size_t)sg * (B_ * D_) + idx];
        ci += pim[(size_t)sg * (B_ * D_) + idx];
    }
    const int row  = idx >> 13;                           // / D_
    const int kout = (idx >> 9) & 15;                     // / DR_ mod K_
    out[idx] = (cr * cr + ci * ci) * wbuf[row * K_ + kout];
}

extern "C" void kernel_launch(void* const* d_in, const int* in_sizes, int n_in,
                              void* d_out, int out_size, void* d_ws, size_t ws_size,
                              hipStream_t stream) {
    const float* psi_re = (const float*)d_in[0];
    const float* psi_im = (const float*)d_in[1];
    const float* U0_re  = (const float*)d_in[2];
    const float* U0_im  = (const float*)d_in[3];
    const float* V_re   = (const float*)d_in[4];
    const float* V_im   = (const float*)d_in[5];
    const int*   eids   = (const int*)d_in[6];
    float* out = (float*)d_out;

    const size_t NBD = (size_t)B_ * D_;
    auto need = [&](int s1, int s2) -> size_t {
        return 4 * NBD * sizeof(__bf16)                       // psi_bf + a_n
             + 2 * (size_t)s1 * NBD * sizeof(float)           // gemm1 partials
             + 2 * (size_t)s2 * NBD * sizeof(float)           // gemm2 partials
             + 1024 * sizeof(float);                          // w
    };
    int S1 = 8, S2 = 4;                                       // split-K factors
    if (need(S1, S2) > ws_size) { S1 = 4; S2 = 2; }
    if (need(S1, S2) > ws_size) { S1 = 2; S2 = 1; }
    if (need(S1, S2) > ws_size) { S1 = 1; S2 = 1; }

    char* p = (char*)d_ws;
    __bf16* psi_bre = (__bf16*)p;  p += NBD * sizeof(__bf16);
    __bf16* psi_bim = (__bf16*)p;  p += NBD * sizeof(__bf16);
    float*  p1re    = (float*)p;   p += (size_t)S1 * NBD * sizeof(float);
    float*  p1im    = (float*)p;   p += (size_t)S1 * NBD * sizeof(float);
    __bf16* anre    = (__bf16*)p;  p += NBD * sizeof(__bf16);
    __bf16* anim    = (__bf16*)p;  p += NBD * sizeof(__bf16);
    float*  p2re    = (float*)p;   p += (size_t)S2 * NBD * sizeof(float);
    float*  p2im    = (float*)p;   p += (size_t)S2 * NBD * sizeof(float);
    float*  wbuf    = (float*)p;

    k_normalize<<<64, 256, 0, stream>>>(psi_re, psi_im, psi_bre, psi_bim);
    k_gemm1<<<dim3(32, S1), 512, 0, stream>>>(psi_bre, psi_bim, U0_re, U0_im,
                                              p1re, p1im, D_ / S1);
    k_stats<<<128, 256, 0, stream>>>(p1re, p1im, S1, wbuf, anre, anim);
    k_gemm2<<<dim3(32, S2), 512, 0, stream>>>(anre, anim, V_re, V_im, eids,
                                              p2re, p2im, DR_ / S2);
    k_out<<<2048, 256, 0, stream>>>(p2re, p2im, S2, wbuf, out);
}
